// SAT_53953379173094
// MI455X (gfx1250) — compile-verified, matrix operand order checked by reading, not run on hardware
//
#include <hip/hip_runtime.h>
#include <hip/hip_bf16.h>
#include <stdint.h>

#define NNODES 50000
#define NEDGES 400000
#define DIM    512
#define INNER  2048
#define BN_EPS 1e-5f
#define CH     4          // K-tiles (of 32) staged per async chunk

typedef __attribute__((ext_vector_type(16))) __bf16 v16bf;
typedef __attribute__((ext_vector_type(8)))  float  v8f;

union ABu { v16bf v; uint4 u[2]; };

__device__ __forceinline__ unsigned short f2bf(float f) {
  unsigned u = __float_as_uint(f);
  u += 0x7FFFu + ((u >> 16) & 1u);          // round-to-nearest-even
  return (unsigned short)(u >> 16);
}
__device__ __forceinline__ float bf2f(unsigned short h) {
  return __uint_as_float(((unsigned)h) << 16);
}

// ---------------------------------------------------------------- degree
__global__ void deg_count_kernel(const long long* __restrict__ ei, float* __restrict__ deg) {
  int t = blockIdx.x * blockDim.x + threadIdx.x;
  if (t >= NEDGES) return;
  atomicAdd(&deg[(int)ei[NEDGES + t]], 1.0f);           // dst row of edge_index
}
__global__ void dinv_fin_kernel(float* __restrict__ deg) {
  int t = blockIdx.x * blockDim.x + threadIdx.x;
  if (t >= NNODES) return;
  deg[t] = rsqrtf(deg[t] + 1.0f);                       // in-degree + self loop
}

// ------------------------------------------------- pack weights to B-fragment order
// layout: element index = ((ntile*ktiles + ktile)*32 + lane)*16 + j
//   col  n = ntile*16 + (lane&15)
//   k      = ktile*32 + (lane>>4)*8 + (j<8 ? j : j+8)
__global__ void pack_w_kernel(const float* __restrict__ src, unsigned short* __restrict__ dst,
                              int ktiles, int ncols, const float* __restrict__ scale,
                              long long total) {
  long long t = (long long)blockIdx.x * blockDim.x + threadIdx.x;
  if (t >= total) return;
  int j    = (int)(t & 15);
  int lane = (int)((t >> 4) & 31);
  long long q = t >> 9;
  int kt  = (int)(q % ktiles);
  int ntg = (int)(q / ktiles);
  int n = ntg * 16 + (lane & 15);
  int k = kt * 32 + ((lane >> 4) << 3) + (j < 8 ? j : j + 8);
  float v = src[(size_t)k * ncols + n];
  if (scale) v *= scale[k];
  dst[t] = f2bf(v);
}

// ------------------------------------------------- copy x into bf16 concat section 0
__global__ void pack_x_kernel(const float* __restrict__ x, unsigned short* __restrict__ xc) {
  size_t t = (size_t)blockIdx.x * blockDim.x + threadIdx.x;
  if (t >= (size_t)NNODES * DIM) return;
  size_t i = t / DIM, d = t % DIM;
  xc[i * INNER + d] = f2bf(x[t]);
}

// ------------------------------------------------- WMMA GEMM: C[M,ldc] = A_bf16 @ Bpack (+bias)
// One wave -> 16 rows x 64 cols, 8 waves/block. B tile staged into LDS with
// double-buffered global_load_async_to_lds_b128 (ASYNCcnt), A streamed from global.
__global__ __launch_bounds__(256) void gemm_bf16_wmma(
    const unsigned short* __restrict__ A, int lda, int acol,
    const unsigned short* __restrict__ Bp, int ktiles,
    float* __restrict__ C, int ldc, const float* __restrict__ bias, int mtiles) {
  __shared__ __align__(16) unsigned short sB[2][4 * CH * 32 * 16];   // 2 x 16 KB

  const int tid  = threadIdx.x;
  const int wave = tid >> 5;
  const int lane = tid & 31;
  const int mt   = blockIdx.x * 8 + wave;
  const bool live = (mt < mtiles);
  const int mtc  = live ? mt : (mtiles - 1);       // dummy waves clamp, keep EXEC all-1
  const int m0   = mtc << 4;
  const int n0   = blockIdx.y << 6;
  const int r    = m0 + (lane & 15);
  const int ai   = (lane >> 4);                    // K 0..7 vs 8..15 half select
  const uint4* Ar = (const uint4*)(A + (size_t)r * lda + acol);

  // cooperative async-copy mapping: thread -> (ntile region, 16B slot)
  const int cnt = tid >> 6;                        // 0..3  (ntile within block)
  const int csl = tid & 63;                        // 0..63 (16B slot)
  const char* gbase = (const char*)Bp +
      (size_t)((n0 >> 4) + cnt) * ktiles * 1024 + (size_t)csl * 16;
  const unsigned ldsreg = (unsigned)(size_t)(void*)&sB[0][0] +
                          (unsigned)cnt * (CH * 1024) + (unsigned)csl * 16;

  const int numc = ktiles / CH;

  // prologue: async-stage chunk 0 into buffer 0
  #pragma unroll
  for (int i = 0; i < CH; ++i) {
    unsigned long long ga = (unsigned long long)(size_t)(gbase + i * 1024);
    unsigned la = ldsreg + (unsigned)(i * 1024);
    asm volatile("global_load_async_to_lds_b128 %0, %1, off" :: "v"(la), "v"(ga) : "memory");
  }

  v8f acc0 = (v8f)0.0f, acc1 = (v8f)0.0f, acc2 = (v8f)0.0f, acc3 = (v8f)0.0f;

  for (int c = 0; c < numc; ++c) {
    if (c + 1 < numc) {
      // issue next chunk into the other buffer, keep it in flight
      const char* g = gbase + (size_t)(c + 1) * (CH * 1024);
      unsigned l = ldsreg + (unsigned)((c + 1) & 1) * (4 * CH * 1024);
      __builtin_prefetch(Ar + (c + 1) * CH * 4, 0, 0);      // prefetch next A chunk
      #pragma unroll
      for (int i = 0; i < CH; ++i) {
        unsigned long long ga = (unsigned long long)(size_t)(g + i * 1024);
        unsigned la = l + (unsigned)(i * 1024);
        asm volatile("global_load_async_to_lds_b128 %0, %1, off" :: "v"(la), "v"(ga) : "memory");
      }
      asm volatile("s_wait_asynccnt 0x4" ::: "memory");      // chunk c landed
    } else {
      asm volatile("s_wait_asynccnt 0x0" ::: "memory");
    }
    __syncthreads();                                         // all waves' fills visible

    const unsigned short* sb = &sB[c & 1][0];
    #pragma unroll
    for (int ktl = 0; ktl < CH; ++ktl) {
      const int kt = c * CH + ktl;
      ABu a;
      a.u[0] = Ar[kt * 4 + ai];                              // K = kbase .. kbase+7
      a.u[1] = Ar[kt * 4 + ai + 2];                          // K = kbase+16 .. kbase+23
      const uint4* bq = (const uint4*)sb + ((size_t)ktl * 32 + lane) * 2;
      const size_t ns = (size_t)CH * 64;                     // per-ntile stride in uint4
      ABu b0, b1, b2, b3;
      b0.u[0] = bq[0];          b0.u[1] = bq[1];
      b1.u[0] = bq[ns];         b1.u[1] = bq[ns + 1];
      b2.u[0] = bq[2 * ns];     b2.u[1] = bq[2 * ns + 1];
      b3.u[0] = bq[3 * ns];     b3.u[1] = bq[3 * ns + 1];
      acc0 = __builtin_amdgcn_wmma_f32_16x16x32_bf16(false, a.v, false, b0.v, (short)0, acc0, false, false);
      acc1 = __builtin_amdgcn_wmma_f32_16x16x32_bf16(false, a.v, false, b1.v, (short)0, acc1, false, false);
      acc2 = __builtin_amdgcn_wmma_f32_16x16x32_bf16(false, a.v, false, b2.v, (short)0, acc2, false, false);
      acc3 = __builtin_amdgcn_wmma_f32_16x16x32_bf16(false, a.v, false, b3.v, (short)0, acc3, false, false);
    }
    __syncthreads();                                         // buffer safe to overwrite
  }

  if (live) {
    const int mr = m0 + ((lane >> 4) << 3);
    v8f accs[4] = {acc0, acc1, acc2, acc3};
    #pragma unroll
    for (int nt = 0; nt < 4; ++nt) {
      const int col = n0 + nt * 16 + (lane & 15);
      const float bi = bias ? bias[col] : 0.0f;
      #pragma unroll
      for (int v = 0; v < 8; ++v)
        C[(size_t)(mr + v) * ldc + col] = accs[nt][v] + bi;
    }
  }
}

// ------------------------------------------------- edge scatter-aggregate
__global__ void edge_agg_kernel(const long long* __restrict__ ei, const float* __restrict__ h,
                                const float* __restrict__ dinv, float* __restrict__ agg) {
  const int e = blockIdx.x;
  const int s = (int)ei[e];
  const int d = (int)ei[NEDGES + e];
  const float nrm = dinv[s] * dinv[d];
  const float* hs = h + (size_t)s * DIM;
  float* ad = agg + (size_t)d * DIM;
  for (int c = threadIdx.x; c < DIM; c += blockDim.x)
    atomicAdd(&ad[c], hs[c] * nrm);
}

// ------------------------------------------------- self-loop + bias + GELU -> bf16 concat
__global__ void post_kernel(const float* __restrict__ agg, const float* __restrict__ h,
                            const float* __restrict__ dinv, const float* __restrict__ b,
                            unsigned short* __restrict__ xc, int sec) {
  size_t t = (size_t)blockIdx.x * blockDim.x + threadIdx.x;
  if (t >= (size_t)NNODES * DIM) return;
  size_t i = t / DIM, d = t % DIM;
  float di = dinv[i];
  float v = agg[t] + h[t] * di * di + b[d];
  float g = 0.5f * v * (1.0f + erff(v * 0.70710678118654752440f));
  xc[i * INNER + (size_t)sec * DIM + d] = f2bf(g);
}

// ------------------------------------------------- BatchNorm column statistics
__global__ void stats_partial_kernel(const unsigned short* __restrict__ xc,
                                     float* __restrict__ csum, float* __restrict__ csq) {
  const int r0 = blockIdx.x * 250;                // 200 groups * 250 rows = 50000
  for (int c = threadIdx.x; c < INNER; c += blockDim.x) {
    float s = 0.f, q = 0.f;
    const unsigned short* p = xc + (size_t)r0 * INNER + c;
    for (int r = 0; r < 250; ++r) {
      float v = bf2f(p[(size_t)r * INNER]);
      s += v; q += v * v;
    }
    atomicAdd(&csum[c], s);
    atomicAdd(&csq[c], q);
  }
}
__global__ void stats_final_kernel(const float* __restrict__ csum, const float* __restrict__ csq,
                                   const float* __restrict__ gamma, const float* __restrict__ beta,
                                   float* __restrict__ sbuf, float* __restrict__ fbuf) {
  int c = blockIdx.x * blockDim.x + threadIdx.x;
  if (c >= INNER) return;
  float mean = csum[c] * (1.0f / NNODES);
  float var  = csq[c] * (1.0f / NNODES) - mean * mean;
  float s = gamma[c] * rsqrtf(var + BN_EPS);
  sbuf[c] = s;                                    // folded into out_W rows
  fbuf[c] = beta[c] - mean * s;                   // folded into output bias
}

// ------------------------------------------------- folded output bias: t = fbuf @ out_W + out_b
__global__ void tvec_kernel(const float* __restrict__ fbuf, const float* __restrict__ outW,
                            const float* __restrict__ outb, float* __restrict__ tvec) {
  __shared__ float red[256];
  const int n = blockIdx.x;
  float s = 0.f;
  for (int k = threadIdx.x; k < INNER; k += 256)
    s += fbuf[k] * outW[(size_t)k * DIM + n];
  red[threadIdx.x] = s;
  __syncthreads();
  for (int off = 128; off > 0; off >>= 1) {
    if ((int)threadIdx.x < off) red[threadIdx.x] += red[threadIdx.x + off];
    __syncthreads();
  }
  if (threadIdx.x == 0) tvec[n] = red[0] + outb[n];
}

// ================================================================ launch
extern "C" void kernel_launch(void* const* d_in, const int* in_sizes, int n_in,
                              void* d_out, int out_size, void* d_ws, size_t ws_size,
                              hipStream_t stream) {
  const float*     x     = (const float*)d_in[0];
  const long long* ei    = (const long long*)d_in[1];
  const float*     Ws    = (const float*)d_in[2];
  const float*     bs    = (const float*)d_in[3];
  const float*     gamma = (const float*)d_in[4];
  const float*     beta  = (const float*)d_in[5];
  const float*     outW  = (const float*)d_in[6];
  const float*     outb  = (const float*)d_in[7];
  float*           out   = (float*)d_out;

  char* w = (char*)d_ws;
  unsigned short* xc   = (unsigned short*)(w);                 // N*INNER bf16   (200.0 MB)
  float*          hlin = (float*)(w + 204800000);              // N*DIM f32      (100.0 MB)
  float*          agg  = (float*)(w + 307200000);              // N*DIM f32      (100.0 MB)
  float*          dinv = (float*)(w + 409600000);              // N f32 (deg then dinv)
  unsigned short* wp3  = (unsigned short*)(w + 409800192);     // 3*D*D bf16 packed
  unsigned short* wpo  = (unsigned short*)(w + 411373056);     // INNER*D bf16 packed
  float*          csum = (float*)(w + 413470208);
  float*          csq  = (float*)(w + 413478400);
  float*          sbuf = (float*)(w + 413486592);
  float*          fbuf = (float*)(w + 413494784);
  float*          tv   = (float*)(w + 413502976);

  // degree -> dinv
  hipMemsetAsync(dinv, 0, NNODES * sizeof(float), stream);
  deg_count_kernel<<<(NEDGES + 255) / 256, 256, 0, stream>>>(ei, dinv);
  dinv_fin_kernel<<<(NNODES + 255) / 256, 256, 0, stream>>>(dinv);

  // pack layer weights and input features
  for (int l = 0; l < 3; ++l)
    pack_w_kernel<<<(262144 + 255) / 256, 256, 0, stream>>>(Ws + (size_t)l * 262144,
                                                            wp3 + (size_t)l * 262144,
                                                            16, DIM, nullptr, 262144LL);
  pack_x_kernel<<<((NNODES * DIM) + 255) / 256, 256, 0, stream>>>(x, xc);

  const int MT = NNODES / 16;                     // 3125 wave tiles
  dim3 gg((MT + 7) / 8, DIM / 64);                // (391, 8)

  // GCN layers
  for (int l = 0; l < 3; ++l) {
    gemm_bf16_wmma<<<gg, 256, 0, stream>>>(xc, INNER, l * DIM,
                                           wp3 + (size_t)l * 262144, 16,
                                           hlin, DIM, nullptr, MT);
    hipMemsetAsync(agg, 0, (size_t)NNODES * DIM * sizeof(float), stream);
    edge_agg_kernel<<<NEDGES, 256, 0, stream>>>(ei, hlin, dinv, agg);
    post_kernel<<<((NNODES * DIM) + 255) / 256, 256, 0, stream>>>(agg, hlin, dinv,
                                                                  bs + (size_t)l * DIM,
                                                                  xc, l + 1);
  }

  // BatchNorm statistics + fold into final GEMM
  hipMemsetAsync(csum, 0, INNER * sizeof(float), stream);
  hipMemsetAsync(csq, 0, INNER * sizeof(float), stream);
  stats_partial_kernel<<<200, 256, 0, stream>>>(xc, csum, csq);
  stats_final_kernel<<<(INNER + 255) / 256, 256, 0, stream>>>(csum, csq, gamma, beta, sbuf, fbuf);
  pack_w_kernel<<<((INNER * DIM) + 255) / 256, 256, 0, stream>>>(outW, wpo, INNER / 32, DIM, sbuf,
                                                                 (long long)INNER * DIM);
  tvec_kernel<<<DIM, 256, 0, stream>>>(fbuf, outW, outb, tv);

  // final GEMM: out = xc @ (diag(s) out_W) + t
  gemm_bf16_wmma<<<gg, 256, 0, stream>>>(xc, INNER, 0, wpo, INNER / 32, out, DIM, tv, MT);
}
